// LinearChunk_54820962566193
// MI455X (gfx1250) — matrix operand order repaired and checked by reading, not compiled, not run on hardware
//
#include <hip/hip_runtime.h>
#include <hip/hip_bf16.h>

typedef __attribute__((ext_vector_type(16))) _Float16 v16h;
typedef __attribute__((ext_vector_type(8)))  _Float16 v8h;
typedef __attribute__((ext_vector_type(8)))  float    v8f;
typedef __attribute__((ext_vector_type(4)))  float    v4f;

#define B_SZ  2048
#define K_SZ  128
#define D_IN  256
#define FEAT  768            // 3 * D_IN
#define XPAD  (FEAT + D_IN)  // extra zeroed 256-entry region for dead columns

// One block per batch row b (256 threads = 8 waves).
// Each wave handles a 16-label tile: C[m,h] = <weight[label_m], x[b, hop h]>
// via 8x v_wmma_f32_16x16x32_f16, then combines with softmax(att) + bias.
__global__ __launch_bounds__(256) void linear_chunk_wmma(
    const float* __restrict__ x,          // (B, 768)
    const long long* __restrict__ labels, // (B, K) int64
    const float* __restrict__ weight,     // (N_LABELS, 256)
    const float* __restrict__ bias_p,     // (N_LABELS,)
    const float* __restrict__ att,        // (N_LABELS, 3)
    float* __restrict__ out)              // (B, K)
{
    __shared__ alignas(16) _Float16 xs[XPAD];  // x[b,:] as f16 + zero pad
    __shared__ float cdump[8][16][4];          // per-wave C columns 0..2

    const int b      = blockIdx.x;
    const int tid    = threadIdx.x;
    const int wave   = tid >> 5;
    const int lane   = tid & 31;
    const int laneLo = lane & 15;
    const int hiOff8  = (lane & 16) ? 8  : 0;   // A-fragment K offset, upper half-wave
    const int hiOff16 = (lane & 16) ? 8 : 0;    // B-fragment K offset, upper half-wave

    // ---- stage x[b,:] -> LDS (f16), plus zeroed pad region, once per block ----
    for (int i = tid; i < XPAD; i += 256)
        xs[i] = (i < FEAT) ? (_Float16)x[(long long)b * FEAT + i] : (_Float16)0.0f;
    __syncthreads();

    // ---- per-label metadata (lane pair L/L+16 redundant) ----
    const int k0 = wave * 16;
    const long long lab = labels[(long long)b * K_SZ + k0 + laneLo];
    const float* __restrict__ wrow = weight + lab * (long long)D_IN;

    const float a0 = att[lab * 3 + 0];
    const float a1 = att[lab * 3 + 1];
    const float a2 = att[lab * 3 + 2];
    const float mx = fmaxf(a0, fmaxf(a1, a2));
    const float e0 = __expf(a0 - mx), e1 = __expf(a1 - mx), e2 = __expf(a2 - mx);
    const float inv = 1.0f / (e0 + e1 + e2);
    const float s0 = e0 * inv, s1 = e1 * inv, s2 = e2 * inv;

    v8f c = {};              // f32 accumulator tile
    const int hop  = laneLo; // B-matrix column index == hop (only 0..2 live)
    // dead columns (hop >= 3) read from the zeroed pad -> branchless B fragment
    const int xoff = (hop < 3) ? hop * D_IN : FEAT;

    #pragma unroll
    for (int s = 0; s < 8; ++s) {
        // ---- A: 16 f32 weights -> f16 fragment (rows = gathered labels) ----
        // lane pair (L, L+16) together covers one contiguous 128B line per step
        const int abase = s * 32 + hiOff8;
        const v4f w0 = *(const v4f*)(wrow + abase + 0);
        const v4f w1 = *(const v4f*)(wrow + abase + 4);
        const v4f w2 = *(const v4f*)(wrow + abase + 16);
        const v4f w3 = *(const v4f*)(wrow + abase + 20);
        v16h a;
        a[0]=(_Float16)w0[0];  a[1]=(_Float16)w0[1];  a[2]=(_Float16)w0[2];  a[3]=(_Float16)w0[3];
        a[4]=(_Float16)w1[0];  a[5]=(_Float16)w1[1];  a[6]=(_Float16)w1[2];  a[7]=(_Float16)w1[3];
        a[8]=(_Float16)w2[0];  a[9]=(_Float16)w2[1];  a[10]=(_Float16)w2[2]; a[11]=(_Float16)w2[3];
        a[12]=(_Float16)w3[0]; a[13]=(_Float16)w3[1]; a[14]=(_Float16)w3[2]; a[15]=(_Float16)w3[3];

        // ---- B: column h = x[b, h*256 + k]; columns 3..15 read zeros ----
        const _Float16* xp = xs + xoff + s * 32 + hiOff16;
        const v8h x0 = *(const v8h*)(xp);
        const v8h x1 = *(const v8h*)(xp + 16);
        v16h bm;
        bm[0]=x0[0]; bm[1]=x0[1]; bm[2]=x0[2];  bm[3]=x0[3];
        bm[4]=x0[4]; bm[5]=x0[5]; bm[6]=x0[6];  bm[7]=x0[7];
        bm[8]=x1[0]; bm[9]=x1[1]; bm[10]=x1[2]; bm[11]=x1[3];
        bm[12]=x1[4];bm[13]=x1[5];bm[14]=x1[6]; bm[15]=x1[7];

        // D = A*B + C  (EXEC all-ones; no divergence in the loop)
        c = __builtin_amdgcn_wmma_f32_16x16x32_f16(
                /*neg_a=*/false, a, /*neg_b=*/false, bm,
                /*c_mod=*/(short)0, c, /*reuse_a=*/false, /*reuse_b=*/false);
    }

    // ---- extract columns 0..2 of C: lane n<3 holds (M=i, N=n) in c[i] (lo half),
    //      lane 16+n holds (M=i+8, N=n) ----
    if (laneLo < 3) {
        const int mbase = (lane & 16) ? 8 : 0;
        #pragma unroll
        for (int i = 0; i < 8; ++i)
            cdump[wave][mbase + i][laneLo] = c[i];
    }
    __syncthreads();

    // ---- lane m: out = sum_h softmax_h * dot_h + bias ----
    if (lane < 16) {
        const float r = s0 * cdump[wave][lane][0]
                      + s1 * cdump[wave][lane][1]
                      + s2 * cdump[wave][lane][2]
                      + bias_p[lab];
        out[(long long)b * K_SZ + k0 + lane] = r;
    }
}

extern "C" void kernel_launch(void* const* d_in, const int* in_sizes, int n_in,
                              void* d_out, int out_size, void* d_ws, size_t ws_size,
                              hipStream_t stream) {
    const float*     x      = (const float*)d_in[0];
    const long long* labels = (const long long*)d_in[1];  // int64 per reference
    const float*     weight = (const float*)d_in[2];
    const float*     bias_p = (const float*)d_in[3];
    const float*     att    = (const float*)d_in[4];
    float*           out    = (float*)d_out;

    linear_chunk_wmma<<<dim3(B_SZ), dim3(256), 0, stream>>>(
        x, labels, weight, bias_p, att, out);
}
